// RNN_72000831750677
// MI455X (gfx1250) — matrix operand order repaired and checked by reading, not compiled, not run on hardware
//
#include <hip/hip_runtime.h>

// Problem constants (reference: B=64, T=2048, D=256, H=256, fp32)
#define Bsz 64
#define Tsz 2048
#define Dsz 256
#define Hsz 256

typedef __attribute__((ext_vector_type(16))) _Float16 v16h;
typedef __attribute__((ext_vector_type(8)))  _Float16 f16x8;
typedef __attribute__((ext_vector_type(4)))  _Float16 f16x4;
typedef __attribute__((ext_vector_type(8)))  float    v8f;
typedef __attribute__((ext_vector_type(4)))  float    f32x4;
typedef __attribute__((ext_vector_type(4)))  int      i32x4;

// Pointer-with-address-space types for the async copy builtin
// (clang diagnostic showed param0 = "int4 __device__ *", i.e. addrspace(1)).
typedef __attribute__((address_space(1))) i32x4 as1_i32x4;
typedef __attribute__((address_space(3))) i32x4 as3_i32x4;

#define WMMA_F16(a, b, c) \
  __builtin_amdgcn_wmma_f32_16x16x32_f16(false, (a), false, (b), (short)0, (c), false, false)

// CDNA5 async global->LDS copy path (ASYNCcnt-tracked, bypasses VGPRs).
#if __has_builtin(__builtin_amdgcn_global_load_async_to_lds_b128) && \
    __has_builtin(__builtin_amdgcn_s_wait_asynccnt)
#define HAVE_ASYNC_LDS 1
#else
#define HAVE_ASYNC_LDS 0
#endif

// A-fragment (16x32 f16) from a row-major f16 buffer (LDS), ld = ldw halves.
// ISA layout: lanes 0-15: M=lane, halves = K{k0..k0+7, k0+16..k0+23};
//             lanes 16-31: M=lane-16, halves = K{k0+8..k0+15, k0+24..k0+31}.
__device__ __forceinline__ v16h load_a_frag_lds(const _Float16* A, int ldw, int lane, int k0) {
  int row = lane & 15;
  const _Float16* p = A + row * ldw + k0 + ((lane >> 4) << 3);
  f16x8 lo = *(const f16x8*)(p);
  f16x8 hi = *(const f16x8*)(p + 16);
  v16h r;
#pragma unroll
  for (int i = 0; i < 8; i++) { r[i] = lo[i]; r[i + 8] = hi[i]; }
  return r;
}

// Same A-fragment, but from a row-major fp32 buffer (LDS, async-copied raw),
// converting to f16 on the fly (co-issues under the WMMAs). ld = ldw floats.
__device__ __forceinline__ v16h load_a_frag_lds_f32(const float* A, int ldw, int lane, int k0) {
  int row = lane & 15;
  const float* p = A + row * ldw + k0 + ((lane >> 4) << 3);
  f32x4 a0 = *(const f32x4*)(p);
  f32x4 a1 = *(const f32x4*)(p + 4);
  f32x4 b0 = *(const f32x4*)(p + 16);
  f32x4 b1 = *(const f32x4*)(p + 20);
  v16h r;
#pragma unroll
  for (int i = 0; i < 4; i++) {
    r[i]      = (_Float16)a0[i];
    r[i + 4]  = (_Float16)a1[i];
    r[i + 8]  = (_Float16)b0[i];
    r[i + 12] = (_Float16)b1[i];
  }
  return r;
}

// B-fragment (32x16 f16) from a COLUMN-major (transposed, [N][K], ldk=256)
// f16 weight matrix in global memory.
// ISA layout: lanes 0-15: N=n0+lane, halves = K{k0..k0+7, k0+16..k0+23};
//             lanes 16-31: N=n0+lane-16, halves = K{k0+8..k0+15, k0+24..k0+31}.
__device__ __forceinline__ v16h load_b_frag_g(const _Float16* Wt, int lane, int k0, int n0) {
  int n = n0 + (lane & 15);
  const _Float16* p = Wt + (size_t)n * 256 + k0 + ((lane >> 4) << 3);
  f16x8 lo = *(const f16x8*)(p);
  f16x8 hi = *(const f16x8*)(p + 16);
  v16h r;
#pragma unroll
  for (int i = 0; i < 8; i++) { r[i] = lo[i]; r[i + 8] = hi[i]; }
  return r;
}

// tanh(x) = sign(x) * (1 - e^{-2|x|}) / (1 + e^{-2|x|}); two trans ops (exp2, rcp).
__device__ __forceinline__ float fast_tanh(float x) {
  float ax = __builtin_fabsf(x);
  float t  = __builtin_amdgcn_exp2f(ax * -2.88539008177792681472f);  // -2*log2(e)
  float r  = (1.0f - t) * __builtin_amdgcn_rcpf(1.0f + t);
  return __builtin_copysignf(r, x);
}

// ---------------------------------------------------------------------------
// Kernel 0: transpose-convert both weight matrices fp32[K][N] -> f16[N][K].
// ---------------------------------------------------------------------------
__global__ void wt_prep(const float* __restrict__ Wxh, const float* __restrict__ Whh,
                        _Float16* __restrict__ WxhT, _Float16* __restrict__ WhhT) {
  int n = blockIdx.x;
  int k = threadIdx.x;
  const float* src = blockIdx.y ? Whh : Wxh;
  _Float16*    dst = blockIdx.y ? WhhT : WxhT;
  dst[n * 256 + k] = (_Float16)src[k * 256 + n];
}

// ---------------------------------------------------------------------------
// Kernel 1: xp = x @ Wxh + b  -> written into d_out (in-place staging of xp).
// Block = 256 threads (8 waves), covers 32 rows x 256 cols.
// x tile staged into LDS with GLOBAL_LOAD_ASYNC_TO_LDS_B128 when available.
// ---------------------------------------------------------------------------
__global__ __launch_bounds__(256) void xproj(const float* __restrict__ x,
                                             const _Float16* __restrict__ WxhT,
                                             const float* __restrict__ bias,
                                             float* __restrict__ out) {
  const int tid  = threadIdx.x;
  const int wave = tid >> 5;
  const int lane = tid & 31;
  const int gm0  = blockIdx.x * 32;

#if HAVE_ASYNC_LDS
  // Raw fp32 tile, rows padded to 260 dwords (each of 16 A-rows hits a
  // disjoint 4-bank group for the b128 fragment reads).
  __shared__ __align__(16) float Axf[32 * 260];
  {
    const float* gbase = x + (size_t)gm0 * 256;
    for (int j = tid; j < 32 * 64; j += 256) {   // 2048 x 16B transfers
      int rr = j >> 6, seg = (j & 63) << 2;
      __builtin_amdgcn_global_load_async_to_lds_b128(
          (as1_i32x4*)(gbase + rr * 256 + seg),
          (as3_i32x4*)&Axf[rr * 260 + seg],
          0, 0);
    }
    __builtin_amdgcn_s_wait_asynccnt(0);
  }
  __syncthreads();
#else
  // Fallback: vectorized synchronous staging with fp32 -> f16 conversion.
  __shared__ __align__(16) _Float16 Ah[32 * 264];
  for (int i = tid; i < 32 * 64; i += 256) {
    int rr = i >> 6, cc = (i & 63) << 2;
    f32x4 v = *(const f32x4*)&x[(size_t)(gm0 + rr) * 256 + cc];
    f16x4 h;
#pragma unroll
    for (int q = 0; q < 4; q++) h[q] = (_Float16)v[q];
    *(f16x4*)&Ah[rr * 264 + cc] = h;
  }
  __syncthreads();
#endif

  const int n0  = wave * 32;
  const int ln  = lane & 15;
  const int hi8 = (lane >> 4) << 3;

  v8f acc00 = {}, acc01 = {}, acc10 = {}, acc11 = {};
#pragma unroll
  for (int kc = 0; kc < 8; kc++) {
    int k0 = kc * 32;
#if HAVE_ASYNC_LDS
    v16h a0 = load_a_frag_lds_f32(Axf,            260, lane, k0);
    v16h a1 = load_a_frag_lds_f32(Axf + 16 * 260, 260, lane, k0);
#else
    v16h a0 = load_a_frag_lds(Ah,            264, lane, k0);
    v16h a1 = load_a_frag_lds(Ah + 16 * 264, 264, lane, k0);
#endif
    v16h b0 = load_b_frag_g(WxhT, lane, k0, n0);
    v16h b1 = load_b_frag_g(WxhT, lane, k0, n0 + 16);
    acc00 = WMMA_F16(a0, b0, acc00);
    acc01 = WMMA_F16(a0, b1, acc01);
    acc10 = WMMA_F16(a1, b0, acc10);
    acc11 = WMMA_F16(a1, b1, acc11);
  }

  float b0v = bias[n0 + ln];
  float b1v = bias[n0 + 16 + ln];
#pragma unroll
  for (int r = 0; r < 8; r++) {
    int M = r + hi8;
    size_t o0 = (size_t)(gm0 + M) * 256;
    size_t o1 = (size_t)(gm0 + 16 + M) * 256;
    out[o0 + n0 + ln]      = acc00[r] + b0v;
    out[o0 + n0 + 16 + ln] = acc01[r] + b1v;
    out[o1 + n0 + ln]      = acc10[r] + b0v;
    out[o1 + n0 + 16 + ln] = acc11[r] + b1v;
  }
}

// ---------------------------------------------------------------------------
// Kernel 2: sequential scan h_t = tanh(xp_t + h_{t-1} @ Whh), in-place on out.
// Batch rows independent -> 4 blocks of 16 rows, no inter-block sync.
// Whh B-fragments live in 128 VGPRs for the whole 2048-step scan; h lives in
// LDS as f16 (A operand); xp_t loads straight into the WMMA C accumulator.
// ---------------------------------------------------------------------------
__global__ __launch_bounds__(256) void scan(const _Float16* __restrict__ WhhT,
                                            float* __restrict__ out) {
  __shared__ __align__(16) _Float16 hbuf[16 * 264];
  const int tid  = threadIdx.x;
  const int wave = tid >> 5;
  const int lane = tid & 31;
  const int m0   = blockIdx.x * 16;   // batch rows [m0, m0+16)
  const int n0   = wave * 32;
  const int ln   = lane & 15;
  const int hi8  = (lane >> 4) << 3;

  // Preload this wave's Whh fragments into registers: 2 N-tiles x 8 K-chunks.
  v16h bw0[8], bw1[8];
#pragma unroll
  for (int kc = 0; kc < 8; kc++) {
    bw0[kc] = load_b_frag_g(WhhT, lane, kc * 32, n0);
    bw1[kc] = load_b_frag_g(WhhT, lane, kc * 32, n0 + 16);
  }

  // h_0 = 0
  for (int i = tid; i < 16 * 264; i += 256) hbuf[i] = (_Float16)0.0f;
  __syncthreads();

  for (int t = 0; t < Tsz; t++) {
    // acc <- xp_t (C operand initialized from global; layout matches C/D tile)
    v8f acc0, acc1;
#pragma unroll
    for (int r = 0; r < 8; r++) {
      int M = r + hi8;
      size_t o = ((size_t)(m0 + M) * Tsz + t) * 256;
      acc0[r] = out[o + n0 + ln];
      acc1[r] = out[o + n0 + 16 + ln];
    }
    if (t + 1 < Tsz)  // pull next timestep's xp row toward the WGP
      __builtin_prefetch(&out[((size_t)(m0 + hi8) * Tsz + (t + 1)) * 256 + n0 + ln], 0, 0);

    // acc += h_{t-1} @ Whh
#pragma unroll
    for (int kc = 0; kc < 8; kc++) {
      v16h a = load_a_frag_lds(hbuf, 264, lane, kc * 32);
      acc0 = WMMA_F16(a, bw0[kc], acc0);
      acc1 = WMMA_F16(a, bw1[kc], acc1);
    }
    __syncthreads();  // everyone done reading h_{t-1} from LDS

    // h_t = tanh(acc); write to LDS (next step's A) and to d_out (result)
#pragma unroll
    for (int r = 0; r < 8; r++) {
      int M = r + hi8;
      float h0 = fast_tanh(acc0[r]);
      float h1 = fast_tanh(acc1[r]);
      size_t o = ((size_t)(m0 + M) * Tsz + t) * 256;
      out[o + n0 + ln]      = h0;
      out[o + n0 + 16 + ln] = h1;
      hbuf[M * 264 + n0 + ln]      = (_Float16)h0;
      hbuf[M * 264 + n0 + 16 + ln] = (_Float16)h1;
    }
    __syncthreads();  // h_t visible to all waves before next step
  }
}

extern "C" void kernel_launch(void* const* d_in, const int* in_sizes, int n_in,
                              void* d_out, int out_size, void* d_ws, size_t ws_size,
                              hipStream_t stream) {
  (void)in_sizes; (void)n_in; (void)out_size; (void)ws_size;
  const float* x    = (const float*)d_in[0];   // [B,T,D] fp32
  const float* Wxh  = (const float*)d_in[1];   // [D,H]   fp32
  const float* Whh  = (const float*)d_in[2];   // [H,H]   fp32
  const float* bias = (const float*)d_in[3];   // [H]     fp32
  float* out = (float*)d_out;                  // [B,T,H] fp32

  _Float16* WxhT = (_Float16*)d_ws;            // 128 KB
  _Float16* WhhT = WxhT + 256 * 256;           // 128 KB

  wt_prep<<<dim3(256, 2), 256, 0, stream>>>(Wxh, Whh, WxhT, WhhT);
  xproj<<<(Bsz * Tsz) / 32, 256, 0, stream>>>(x, WxhT, bias, out);
  scan<<<Bsz / 16, 256, 0, stream>>>(WhhT, out);
}
